// JanusModel_10986526343218
// MI455X (gfx1250) — compile-verified, matrix operand order checked by reading, not run on hardware
//
#include <hip/hip_runtime.h>
#include <hip/hip_bf16.h>
#include <math.h>

// ---------------- problem constants ----------------
#define BB   2
#define SS   2048
#define DD   1024
#define HH   16
#define KVHH 4
#define HDD  64

typedef __bf16 v16bf __attribute__((ext_vector_type(16)));
typedef __bf16 v4bf  __attribute__((ext_vector_type(4)));
typedef float  v8f   __attribute__((ext_vector_type(8)));
typedef int    i4v   __attribute__((vector_size(16)));   // int4 for async-LDS builtin

typedef __attribute__((address_space(1))) i4v* g_i4p;    // global int4*
typedef __attribute__((address_space(3))) i4v* l_i4p;    // LDS int4*

#if __has_builtin(__builtin_amdgcn_global_load_async_to_lds_b128) && \
    __has_builtin(__builtin_amdgcn_s_wait_asynccnt)
#define HAVE_ASYNC_LDS 1
#else
#define HAVE_ASYNC_LDS 0
#endif

// 16-byte global -> LDS copy; async DMA (ASYNCcnt) when available.
__device__ __forceinline__ void copy16_g2l(const void* gsrc, void* ldst) {
#if HAVE_ASYNC_LDS
  __builtin_amdgcn_global_load_async_to_lds_b128((g_i4p)gsrc, (l_i4p)ldst,
                                                 /*offset=*/0, /*cpol=*/0);
#else
  *(uint4*)ldst = *(const uint4*)gsrc;
#endif
}
__device__ __forceinline__ void async_wait0() {
#if HAVE_ASYNC_LDS
  __builtin_amdgcn_s_wait_asynccnt(0);
#endif
}

// D = A(16x32 bf16) * B(32x16 bf16) + C(16x16 f32)
__device__ __forceinline__ v8f wmma_bf16(v16bf a, v16bf b, v8f c) {
  return __builtin_amdgcn_wmma_f32_16x16x32_bf16(
      false, a, false, b, (short)0, c, false, false);
}

// 16-bit operand register tile (A-layout; B uses same pattern with lane = n):
// lane holds row (lane&15); elems 0..7 -> k = kb+off.., elems 8..15 -> kb+16+off..,
// off = 8*(lane>=16).   (ISA 7.12.2, 16-bit A 16x32)
__device__ __forceinline__ v16bf load_tile_bf16(const __bf16* base, int ld, int row,
                                                int kb, int lane) {
  const int off = (lane < 16) ? 0 : 8;
  const __bf16* p = base + (size_t)row * ld + kb + off;
  v16bf r;
#pragma unroll
  for (int j = 0; j < 8; ++j) {
    r[j]     = p[j];
    r[j + 8] = p[j + 16];
  }
  return r;
}

// =====================================================================
// Kernel 0: one-shot f32 -> bf16 conversion (x and all weights), so the
// GEMM inner loops move/consume bf16 only (no per-k-step v_cvt work).
// =====================================================================
__global__ __launch_bounds__(256) void cvt_bf16_kernel(const float* __restrict__ in,
                                                       __bf16* __restrict__ out,
                                                       int n4) {
  const int i = blockIdx.x * blockDim.x + threadIdx.x;
  if (i < n4) {
    const float4 v = ((const float4*)in)[i];
    v4bf o;
    o[0] = (__bf16)v.x; o[1] = (__bf16)v.y; o[2] = (__bf16)v.z; o[3] = (__bf16)v.w;
    ((v4bf*)out)[i] = o;
  }
}

// =====================================================================
// Kernel 1: fused QKV projection.  xb(4096x1024 bf16) @ [wq|wk|wv]b^T.
// 8 waves/block share one 64-wide n-tile; bf16 weight k-slab (64x32) is
// async-staged into LDS (double buffered) and shared across the waves.
// Q -> q_ws[b][h][s][d], K -> k_ws[b][kv][s][d], V -> vt_ws[b][kv][d][s].
// =====================================================================
__global__ __launch_bounds__(256) void qkv_kernel(const __bf16* __restrict__ xb,
                                                  const __bf16* __restrict__ wqb,
                                                  const __bf16* __restrict__ wkb,
                                                  const __bf16* __restrict__ wvb,
                                                  __bf16* __restrict__ q_ws,
                                                  __bf16* __restrict__ k_ws,
                                                  __bf16* __restrict__ vt_ws) {
  __shared__ __align__(16) __bf16 wtile[2][64][32];   // 8 KB
  const int tid  = threadIdx.x;
  const int lane = tid & 31;
  const int wave = blockIdx.x * 8 + (tid >> 5);
  const int mt = wave & 255;        // 256 row tiles of 16 (8 consecutive per block)
  const int nt = wave >> 8;         // 24 col tiles of 64 (uniform inside a block)
  const int m0 = mt * 16;
  const int n0 = nt * 64;

  const __bf16* wbase;
  int nloc;
  if (n0 < DD)            { wbase = wqb; nloc = n0; }
  else if (n0 < DD + 256) { wbase = wkb; nloc = n0 - DD; }
  else                    { wbase = wvb; nloc = n0 - DD - 256; }

  // cooperative stage of a 64(row) x 32(k) bf16 slab: 256 x 16B chunks, 1/thread
  auto stage_w = [&](int p, int kb) {
    const int row = tid >> 2, c = tid & 3;     // 4 chunks of 8 bf16 per row
    copy16_g2l(wbase + (size_t)(nloc + row) * DD + kb + c * 8, &wtile[p][row][c * 8]);
  };

  const int am = m0 + (lane & 15);
  v8f acc[4] = {};
  stage_w(0, 0);
  for (int ks = 0; ks < DD / 32; ++ks) {
    const int kb = ks * 32;
    const int p  = ks & 1;
    async_wait0();
    __syncthreads();                         // slab p complete & visible
    if (kb + 32 < DD) stage_w(p ^ 1, kb + 32);
    v16bf a = load_tile_bf16(xb, DD, am, kb, lane);
#pragma unroll
    for (int c = 0; c < 4; ++c) {
      v16bf b = load_tile_bf16(&wtile[p][0][0], 32, c * 16 + (lane & 15), 0, lane);
      acc[c] = wmma_bf16(a, b, acc[c]);
    }
  }

  const int g = lane >> 4;          // C layout: row = m0 + r + 8*g, col = n0 + c*16 + (lane&15)
#pragma unroll
  for (int c = 0; c < 4; ++c) {
#pragma unroll
    for (int r = 0; r < 8; ++r) {
      const int m = m0 + r + 8 * g;
      const int n = n0 + c * 16 + (lane & 15);
      const int b = m / SS, s = m % SS;
      const __bf16 v = (__bf16)acc[c][r];
      if (n < DD) {
        const int h = n >> 6, d = n & 63;
        q_ws[(((size_t)b * HH + h) * SS + s) * HDD + d] = v;
      } else if (n < DD + 256) {
        const int kv = (n - DD) >> 6, d = (n - DD) & 63;
        k_ws[(((size_t)b * KVHH + kv) * SS + s) * HDD + d] = v;
      } else {
        const int kv = (n - DD - 256) >> 6, d = (n - DD - 256) & 63;
        vt_ws[(((size_t)b * KVHH + kv) * HDD + d) * SS + s] = v;   // V transposed
      }
    }
  }
}

// =====================================================================
// Kernel 2: fused flash attention.  Block = 8 waves = 128 q-rows of one
// (b,h).  Per 32-key step the K tile (32x64) and V^T tile (64x32) are
// async-DMA'd to LDS (double buffered, ASYNCcnt) and shared by all waves.
// Masks stream with non-temporal loads.  P re-layout (C->A) via LDS.
// =====================================================================
__global__ __launch_bounds__(256) void attn_kernel(const __bf16* __restrict__ q_ws,
                                                   const __bf16* __restrict__ k_ws,
                                                   const __bf16* __restrict__ vt_ws,
                                                   const float* __restrict__ full_mask,
                                                   const float* __restrict__ tag_mask,
                                                   __bf16* __restrict__ attn_ws) {
  __shared__ __align__(16) __bf16 kbuf[2][32][64];   // [buf][t][d]  8 KB
  __shared__ __align__(16) __bf16 vbuf[2][64][32];   // [buf][d][t]  8 KB
  __shared__ __align__(16) __bf16 plds[8][16][32];   // per-wave P   8 KB
  const int tid  = threadIdx.x;
  const int lane = tid & 31;
  const int wib  = tid >> 5;
  const int blk  = blockIdx.x;          // 512 = B(2) * H(16) * 16 q-blocks
  const int qblk = blk & 15;
  const int h    = (blk >> 4) & 15;
  const int b    = blk >> 8;

  const int kv = (h < 8) ? (h >> 2) : (2 + ((h - 8) >> 2));   // GQA map, rep=4
  const __bf16* Q  = q_ws  + ((size_t)b * HH + h)    * SS * HDD;
  const __bf16* K  = k_ws  + ((size_t)b * KVHH + kv) * SS * HDD;
  const __bf16* VT = vt_ws + ((size_t)b * KVHH + kv) * HDD * SS;
  const float*  MK = ((h < 8) ? full_mask : tag_mask) + (size_t)b * SS * SS;

  const int m0 = qblk * 128 + wib * 16;
  const int g  = lane >> 4;
  const int nl = lane & 15;
  const int off = (lane < 16) ? 0 : 8;

  // cooperative stage of K (32x64, 8 chunks/row) and V^T (64x32, 4 chunks/row)
  auto stage_kv = [&](int p, int t0) {
    { const int t = tid >> 3, c = tid & 7;
      copy16_g2l(K + (size_t)(t0 + t) * HDD + c * 8, &kbuf[p][t][c * 8]); }
    { const int d = tid >> 2, c = tid & 3;
      copy16_g2l(VT + (size_t)d * SS + t0 + c * 8, &vbuf[p][d][c * 8]); }
  };

  // Q A-operands (16x64 split into two K=32 chunks), resident for whole loop.
  const v16bf qa0 = load_tile_bf16(Q, HDD, m0 + nl, 0,  lane);
  const v16bf qa1 = load_tile_bf16(Q, HDD, m0 + nl, 32, lane);

  v8f o[4] = {};
  float mrun[8], lrun[8];
#pragma unroll
  for (int r = 0; r < 8; ++r) { mrun[r] = -1.0e30f; lrun[r] = 0.0f; }

  stage_kv(0, 0);
  for (int i = 0; i < SS / 32; ++i) {
    const int t0 = i * 32;
    const int p  = i & 1;
    async_wait0();
    __syncthreads();                      // buffers p complete & visible
    if (t0 + 32 < SS) stage_kv(p ^ 1, t0 + 32);

    // ---- scores: S = Q.K^T (two 16-col halves, operands from LDS) ----
    v8f s0 = {}, s1 = {};
    {
      v16bf kb0 = load_tile_bf16(&kbuf[p][0][0], HDD, nl,      0,  lane);
      v16bf kb1 = load_tile_bf16(&kbuf[p][0][0], HDD, nl,      32, lane);
      s0 = wmma_bf16(qa0, kb0, s0);
      s0 = wmma_bf16(qa1, kb1, s0);
      v16bf kb2 = load_tile_bf16(&kbuf[p][0][0], HDD, 16 + nl, 0,  lane);
      v16bf kb3 = load_tile_bf16(&kbuf[p][0][0], HDD, 16 + nl, 32, lane);
      s1 = wmma_bf16(qa0, kb2, s1);
      s1 = wmma_bf16(qa1, kb3, s1);
    }
    // ---- mask (NT stream) + online softmax, 16-lane-group reductions ----
#pragma unroll
    for (int r = 0; r < 8; ++r) {
      const int mr = m0 + r + 8 * g;
      const float* mrow = MK + (size_t)mr * SS + t0;
      float x0 = s0[r] * 0.125f + __builtin_nontemporal_load(mrow + nl);
      float x1 = s1[r] * 0.125f + __builtin_nontemporal_load(mrow + 16 + nl);
      float mx = fmaxf(x0, x1);
#pragma unroll
      for (int d = 1; d < 16; d <<= 1) mx = fmaxf(mx, __shfl_xor(mx, d, 32));
      const float mnew = fmaxf(mrun[r], mx);
      const float corr = __expf(mrun[r] - mnew);
      const float p0 = __expf(x0 - mnew);
      const float p1 = __expf(x1 - mnew);
      float rs = p0 + p1;
#pragma unroll
      for (int d = 1; d < 16; d <<= 1) rs += __shfl_xor(rs, d, 32);
      lrun[r] = lrun[r] * corr + rs;
      mrun[r] = mnew;
#pragma unroll
      for (int c = 0; c < 4; ++c) o[c][r] *= corr;
      plds[wib][r + 8 * g][nl]      = (__bf16)p0;
      plds[wib][r + 8 * g][16 + nl] = (__bf16)p1;
    }
    // ---- reload P as A operand (same-wave DS ordering) ----
    v16bf pa;
#pragma unroll
    for (int j = 0; j < 8; ++j) {
      pa[j]     = plds[wib][nl][off + j];
      pa[j + 8] = plds[wib][nl][16 + off + j];
    }
    // ---- O += P * V  (V^T tile in LDS, contiguous along t) ----
#pragma unroll
    for (int c = 0; c < 4; ++c) {
      const __bf16* vrow = &vbuf[p][c * 16 + nl][off];
      v16bf vb;
#pragma unroll
      for (int j = 0; j < 8; ++j) { vb[j] = vrow[j]; vb[j + 8] = vrow[16 + j]; }
      o[c] = wmma_bf16(pa, vb, o[c]);
    }
  }

  // ---- normalize + store attn[b][s][h*64+d] as bf16 for the output GEMM ----
#pragma unroll
  for (int c = 0; c < 4; ++c) {
#pragma unroll
    for (int r = 0; r < 8; ++r) {
      const int mr = m0 + r + 8 * g;
      const float v = o[c][r] / lrun[r];
      attn_ws[((size_t)b * SS + mr) * (HH * HDD) + h * HDD + c * 16 + nl] = (__bf16)v;
    }
  }
}

// =====================================================================
// Kernel 3: output projection.  attn(4096x1024 bf16) @ wob^T -> out f32.
// Same block-shared async bf16 weight staging as kernel 1.
// =====================================================================
__global__ __launch_bounds__(256) void oproj_kernel(const __bf16* __restrict__ attn,
                                                    const __bf16* __restrict__ wob,
                                                    float* __restrict__ out) {
  __shared__ __align__(16) __bf16 wtile[2][64][32];   // 8 KB
  const int tid  = threadIdx.x;
  const int lane = tid & 31;
  const int wave = blockIdx.x * 8 + (tid >> 5);
  const int mt = wave & 255;
  const int nt = wave >> 8;             // 16 n-tiles of 64 (uniform in block)
  const int m0 = mt * 16, n0 = nt * 64;

  auto stage_w = [&](int p, int kb) {
    const int row = tid >> 2, c = tid & 3;
    copy16_g2l(wob + (size_t)(n0 + row) * DD + kb + c * 8, &wtile[p][row][c * 8]);
  };

  v8f acc[4] = {};
  stage_w(0, 0);
  for (int ks = 0; ks < DD / 32; ++ks) {
    const int kb = ks * 32;
    const int p  = ks & 1;
    async_wait0();
    __syncthreads();
    if (kb + 32 < DD) stage_w(p ^ 1, kb + 32);
    v16bf a = load_tile_bf16(attn, DD, m0 + (lane & 15), kb, lane);
#pragma unroll
    for (int c = 0; c < 4; ++c) {
      v16bf b = load_tile_bf16(&wtile[p][0][0], 32, c * 16 + (lane & 15), 0, lane);
      acc[c] = wmma_bf16(a, b, acc[c]);
    }
  }
  const int g = lane >> 4;
#pragma unroll
  for (int c = 0; c < 4; ++c) {
#pragma unroll
    for (int r = 0; r < 8; ++r) {
      const int m = m0 + r + 8 * g;
      out[(size_t)m * DD + n0 + c * 16 + (lane & 15)] = acc[c][r];
    }
  }
}

// =====================================================================
extern "C" void kernel_launch(void* const* d_in, const int* in_sizes, int n_in,
                              void* d_out, int out_size, void* d_ws, size_t ws_size,
                              hipStream_t stream) {
  const float* x         = (const float*)d_in[0];   // (B,S,D)
  const float* full_mask = (const float*)d_in[1];   // (B,1,S,S)
  const float* tag_mask  = (const float*)d_in[2];   // (B,1,S,S)
  const float* wq        = (const float*)d_in[3];   // (1024,1024)
  const float* wk        = (const float*)d_in[4];   // (256,1024)
  const float* wv        = (const float*)d_in[5];   // (256,1024)
  const float* wo        = (const float*)d_in[6];   // (1024,1024)
  float* out             = (float*)d_out;

  // workspace (all bf16):
  //  xb 8MB | wqb 2MB | wkb 0.5MB | wvb 0.5MB | wob 2MB |
  //  q_ws 8MB | k_ws 2MB | vt_ws 2MB | attn_ws 8MB   = 33MB
  char* ws = (char*)d_ws;
  size_t o = 0;
  __bf16* xb      = (__bf16*)(ws + o); o += (size_t)BB * SS * DD * 2;        // 16 MB? no: bf16 -> 8 MB
  __bf16* wqb     = (__bf16*)(ws + o); o += (size_t)DD * DD * 2;
  __bf16* wkb     = (__bf16*)(ws + o); o += (size_t)256 * DD * 2;
  __bf16* wvb     = (__bf16*)(ws + o); o += (size_t)256 * DD * 2;
  __bf16* wob     = (__bf16*)(ws + o); o += (size_t)DD * DD * 2;
  __bf16* q_ws    = (__bf16*)(ws + o); o += (size_t)BB * HH * SS * HDD * 2;
  __bf16* k_ws    = (__bf16*)(ws + o); o += (size_t)BB * KVHH * SS * HDD * 2;
  __bf16* vt_ws   = (__bf16*)(ws + o); o += (size_t)BB * KVHH * HDD * SS * 2;
  __bf16* attn_ws = (__bf16*)(ws + o);

  // K0: one-shot f32 -> bf16 conversions (x + all weights)
  cvt_bf16_kernel<<<(BB * SS * DD / 4 + 255) / 256, 256, 0, stream>>>(x, xb, BB * SS * DD / 4);
  cvt_bf16_kernel<<<(DD * DD / 4 + 255) / 256, 256, 0, stream>>>(wq, wqb, DD * DD / 4);
  cvt_bf16_kernel<<<(256 * DD / 4 + 255) / 256, 256, 0, stream>>>(wk, wkb, 256 * DD / 4);
  cvt_bf16_kernel<<<(256 * DD / 4 + 255) / 256, 256, 0, stream>>>(wv, wvb, 256 * DD / 4);
  cvt_bf16_kernel<<<(DD * DD / 4 + 255) / 256, 256, 0, stream>>>(wo, wob, DD * DD / 4);

  // K1: 256 m-tiles * 24 n-tiles = 6144 waves -> 768 blocks of 8 waves
  qkv_kernel<<<768, 256, 0, stream>>>(xb, wqb, wkb, wvb, q_ws, k_ws, vt_ws);
  // K2: B * H * 16 q-blocks = 512 blocks (8 waves each, 128 q rows/block)
  attn_kernel<<<512, 256, 0, stream>>>(q_ws, k_ws, vt_ws, full_mask, tag_mask, attn_ws);
  // K3: 256 m-tiles * 16 n-tiles = 4096 waves -> 512 blocks
  oproj_kernel<<<512, 256, 0, stream>>>(attn_ws, wob, out);
}
